// tyc_2DGCN_Block_61005715472861
// MI455X (gfx1250) — compile-verified, matrix operand order checked by reading, not run on hardware
//
#include <hip/hip_runtime.h>
#include <hip/hip_bf16.h>

#define DFEAT 64
#define ROWS_PER_BLOCK 128
#define WAVES_PER_BLOCK 8

typedef __attribute__((ext_vector_type(16))) __bf16 v16bf;
typedef __attribute__((ext_vector_type(8)))  float  v8f;

union FragVec { uint4 q[2]; v16bf v; };   // 32 bytes = one WMMA bf16 operand per lane

__device__ __forceinline__ unsigned short f2bf(float f) {
    unsigned int u = __float_as_uint(f);
    unsigned int lsb = (u >> 16) & 1u;           // round-to-nearest-even
    return (unsigned short)((u + 0x7FFFu + lsb) >> 16);
}

// ---------------------------------------------------------------------------
// Pre-swizzle one 64x64 fp32 weight matrix into bf16 WMMA B-fragment layout:
//   swz[frag(kbi,nt)][lane][j]  with j = 2*i+pos,
//   k = kbi*32 + ((i&4)<<2) + ((lane>>4)<<3) + ((i&3)<<1) + pos,  col = nt*16 + (lane&15)
// Output is 8 frags * 32 lanes * 16 bf16 = 8 KB. Thread id == output index.
// ---------------------------------------------------------------------------
__global__ __launch_bounds__(256)
void preswizzle_weights(const float* __restrict__ W, unsigned short* __restrict__ swz)
{
    int t = blockIdx.x * 256 + threadIdx.x;      // 0..4095
    int j    = t & 15;
    int lane = (t >> 4) & 31;
    int frag = t >> 9;                           // 0..7
    int kbi  = frag >> 2;
    int nt   = frag & 3;
    int i    = j >> 1;
    int pos  = j & 1;
    int khalf = (lane >> 4) << 3;
    int col   = nt * 16 + (lane & 15);
    int k = kbi * 32 + ((i & 4) << 2) + khalf + ((i & 3) << 1) + pos;
    swz[t] = f2bf(W[k * DFEAT + col]);
}

// ---------------------------------------------------------------------------
// One wave computes a 16x64 tile of BOTH xw = x@W and res = relu(x@Wr+br).
// All WMMA operands are fragment-native in LDS -> 2x ds_load_b128 each.
// ---------------------------------------------------------------------------
__global__ __launch_bounds__(256)
void gcn_dual_gemm_wmma(const float* __restrict__ x,
                        const uint4* __restrict__ swzW,   // 512 uint4 (8 KB)
                        const uint4* __restrict__ swzWr,  // 512 uint4 (8 KB)
                        const float* __restrict__ brv,
                        float* __restrict__ xw,
                        float* __restrict__ res,
                        int nnodes)
{
    __shared__ uint4 sBW[512];                    // W  fragments (8 KB)
    __shared__ uint4 sBR[512];                    // Wr fragments (8 KB)
    __shared__ uint4 sA[WAVES_PER_BLOCK][128];    // per-wave A fragments (2 KB each)

    const int tid  = threadIdx.x;
    const int wave = tid >> 5;
    const int lane = tid & 31;
    const int m0   = blockIdx.x * ROWS_PER_BLOCK + wave * 16;

    // Stage pre-swizzled weight fragments (block-wide b128 copies).
#pragma unroll
    for (int i = tid; i < 512; i += 256) {
        sBW[i] = swzW[i];
        sBR[i] = swzWr[i];
    }

    // Stage this wave's 16x64 fp32 x-tile, converting straight into A-fragment
    // layout: float4 at (row, k) -> bf16 pairs i0,i0+1 of lane row+16*((k>>3)&1).
#pragma unroll
    for (int it = 0; it < 8; ++it) {
        int idx = it * 32 + lane;                 // 16 rows x 16 float4
        int r   = idx >> 4;
        int c4  = (idx & 15) << 2;                // k base (multiple of 4)
        int m   = m0 + r;
        float4 v = make_float4(0.f, 0.f, 0.f, 0.f);
        if (m < nnodes) v = *reinterpret_cast<const float4*>(x + (long long)m * DFEAT + c4);
        int kbi    = c4 >> 5;
        int kk     = c4 & 31;
        int lane_t = r + (((kk >> 3) & 1) << 4);
        int i0     = ((kk & 16) >> 2) + ((kk & 7) >> 1);   // even: 0,2,4,6
        unsigned lo = (unsigned)f2bf(v.x) | ((unsigned)f2bf(v.y) << 16);
        unsigned hi = (unsigned)f2bf(v.z) | ((unsigned)f2bf(v.w) << 16);
        unsigned* base = reinterpret_cast<unsigned*>(&sA[wave][0]);
        uint2* p = reinterpret_cast<uint2*>(base + (kbi * 32 + lane_t) * 8 + i0);
        *p = make_uint2(lo, hi);                  // one ds_store_b64
    }
    __syncthreads();

    v8f accW[4], accR[4];
#pragma unroll
    for (int nt = 0; nt < 4; ++nt)
#pragma unroll
        for (int r = 0; r < 8; ++r) { accW[nt][r] = 0.f; accR[nt][r] = 0.f; }

#pragma unroll
    for (int kbi = 0; kbi < 2; ++kbi) {
        FragVec a;
        int ai = (kbi * 32 + lane) * 2;
        a.q[0] = sA[wave][ai];
        a.q[1] = sA[wave][ai + 1];
#pragma unroll
        for (int nt = 0; nt < 4; ++nt) {
            FragVec bw, br_;
            int fi = ((kbi * 4 + nt) * 32 + lane) * 2;
            bw.q[0]  = sBW[fi];     bw.q[1]  = sBW[fi + 1];
            br_.q[0] = sBR[fi];     br_.q[1] = sBR[fi + 1];
            accW[nt] = __builtin_amdgcn_wmma_f32_16x16x32_bf16(
                false, a.v, false, bw.v,  (short)0, accW[nt], false, false);
            accR[nt] = __builtin_amdgcn_wmma_f32_16x16x32_bf16(
                false, a.v, false, br_.v, (short)0, accR[nt], false, false);
        }
    }

    // C/D layout: VGPR r, lanes 0-15 -> M=r, lanes 16-31 -> M=r+8, N=lane&15.
    const int col    = lane & 15;
    const int rowoff = (lane >> 4) << 3;
#pragma unroll
    for (int nt = 0; nt < 4; ++nt) {
        float bv = brv[nt * 16 + col];
#pragma unroll
        for (int r = 0; r < 8; ++r) {
            int m = m0 + r + rowoff;
            if (m < nnodes) {
                long long o = (long long)m * DFEAT + nt * 16 + col;
                xw[o] = accW[nt][r];
                float rv = accR[nt][r] + bv;
                res[o] = rv > 0.f ? rv : 0.f;
            }
        }
    }
}

// 16 lanes per edge: float4 gather from xw[src], 4x global_atomic_add_f32 to agg[dst].
__global__ __launch_bounds__(256)
void gcn_scatter(const float4* __restrict__ xw,
                 const int* __restrict__ src,
                 const int* __restrict__ dst,
                 float* __restrict__ agg,
                 int nedges)
{
    long long t = (long long)blockIdx.x * blockDim.x + threadIdx.x;
    int e = (int)(t >> 4);
    int l = (int)(t & 15);
    if (e >= nedges) return;
    int s = src[e];
    int d = dst[e];
    float4 v = xw[(long long)s * 16 + l];
    float* o = agg + (long long)d * DFEAT + l * 4;
    atomicAdd(o + 0, v.x);
    atomicAdd(o + 1, v.y);
    atomicAdd(o + 2, v.z);
    atomicAdd(o + 3, v.w);
}

__global__ __launch_bounds__(256)
void zero4(float4* __restrict__ p, long long n4)
{
    long long i = (long long)blockIdx.x * blockDim.x + threadIdx.x;
    if (i < n4) p[i] = make_float4(0.f, 0.f, 0.f, 0.f);
}

// out = relu(agg + b) + res   (res already relu'ed residual branch)
__global__ __launch_bounds__(256)
void gcn_finalize(const float4* __restrict__ agg,
                  const float* __restrict__ b,
                  const float4* __restrict__ res,
                  float4* __restrict__ out,
                  long long n4)
{
    long long i = (long long)blockIdx.x * blockDim.x + threadIdx.x;
    if (i >= n4) return;
    int c4 = (int)(i & 15);
    float4 bb = reinterpret_cast<const float4*>(b)[c4];
    float4 a = agg[i];
    float4 r = res[i];
    float4 o;
    o.x = fmaxf(a.x + bb.x, 0.f) + r.x;
    o.y = fmaxf(a.y + bb.y, 0.f) + r.y;
    o.z = fmaxf(a.z + bb.z, 0.f) + r.z;
    o.w = fmaxf(a.w + bb.w, 0.f) + r.w;
    out[i] = o;
}

extern "C" void kernel_launch(void* const* d_in, const int* in_sizes, int n_in,
                              void* d_out, int out_size, void* d_ws, size_t ws_size,
                              hipStream_t stream)
{
    const float* feats = (const float*)d_in[0];
    const float* W1    = (const float*)d_in[1];
    const float* b1    = (const float*)d_in[2];
    const float* Wr1   = (const float*)d_in[3];
    const float* br1   = (const float*)d_in[4];
    const float* W2    = (const float*)d_in[5];
    const float* b2    = (const float*)d_in[6];
    const float* Wr2   = (const float*)d_in[7];
    const float* br2   = (const float*)d_in[8];
    const int*   src   = (const int*)d_in[9];
    const int*   dst   = (const int*)d_in[10];
    float*       out   = (float*)d_out;

    const int nnodes = in_sizes[0] / DFEAT;
    const int nedges = in_sizes[9];
    const long long nd = (long long)nnodes * DFEAT;
    const long long n4 = nd / 4;

    float* xw  = (float*)d_ws;        // [N,D]
    float* res = xw + nd;             // [N,D]
    float* agg = res + nd;            // [N,D]
    unsigned short* swz = (unsigned short*)(agg + nd);   // 4 x 4096 bf16 (32 KB)
    unsigned short* sW1  = swz + 0 * 4096;
    unsigned short* sWr1 = swz + 1 * 4096;
    unsigned short* sW2  = swz + 2 * 4096;
    unsigned short* sWr2 = swz + 3 * 4096;

    dim3 gemmGrid((nnodes + ROWS_PER_BLOCK - 1) / ROWS_PER_BLOCK);
    dim3 elwGrid((unsigned)((n4 + 255) / 256));
    long long st = (long long)nedges * 16;
    dim3 scatGrid((unsigned)((st + 255) / 256));

    // Pre-swizzle all four weight matrices into WMMA fragment layout (once).
    preswizzle_weights<<<16, 256, 0, stream>>>(W1,  sW1);
    preswizzle_weights<<<16, 256, 0, stream>>>(Wr1, sWr1);
    preswizzle_weights<<<16, 256, 0, stream>>>(W2,  sW2);
    preswizzle_weights<<<16, 256, 0, stream>>>(Wr2, sWr2);

    // ---- layer 1 (h stored in d_out) ----
    zero4<<<elwGrid, 256, 0, stream>>>((float4*)agg, n4);
    gcn_dual_gemm_wmma<<<gemmGrid, 256, 0, stream>>>(feats, (const uint4*)sW1,
                                                     (const uint4*)sWr1, br1, xw, res, nnodes);
    gcn_scatter<<<scatGrid, 256, 0, stream>>>((const float4*)xw, src, dst, agg, nedges);
    gcn_finalize<<<elwGrid, 256, 0, stream>>>((const float4*)agg, b1, (const float4*)res,
                                              (float4*)out, n4);
    // ---- layer 2 (reads d_out, overwrites d_out) ----
    zero4<<<elwGrid, 256, 0, stream>>>((float4*)agg, n4);
    gcn_dual_gemm_wmma<<<gemmGrid, 256, 0, stream>>>(out, (const uint4*)sW2,
                                                     (const uint4*)sWr2, br2, xw, res, nnodes);
    gcn_scatter<<<scatGrid, 256, 0, stream>>>((const float4*)xw, src, dst, agg, nedges);
    gcn_finalize<<<elwGrid, 256, 0, stream>>>((const float4*)agg, b2, (const float4*)res,
                                              (float4*)out, n4);
}
